// ContrastiveCosineEmbeddingLoss_13950053777737
// MI455X (gfx1250) — matrix operand order repaired and checked by reading, not compiled, not run on hardware
//
#include <hip/hip_runtime.h>

// ---------------------------------------------------------------------------
// ContrastiveCosineEmbeddingLoss, fused for MI455X (gfx1250, wave32, WMMA).
//   cossim = Xn Xn^T computed tile-wise with v_wmma_f32_16x16x32_f16,
//   loss terms folded into the GEMM epilogue (closed-form class weights),
//   only upper-triangular block tiles are processed.
// Wave tile 64x64 (16 WMMA per 16 b128 fragment loads) to maximize
// FLOP/byte out of the WGP$/L2 (whole 4 MB operand set is L2-resident).
// ---------------------------------------------------------------------------

typedef __attribute__((ext_vector_type(16))) _Float16 v16h;
typedef __attribute__((ext_vector_type(8)))  _Float16 v8h;
typedef __attribute__((ext_vector_type(8)))  float    v8f;

#define N_ROWS  8192
#define EDIM    256
#define KSTEP   32
#define MARGIN  0.5f
#define NRM_EPS 1e-8f

struct LossParams {
  float accum;      // global loss accumulator (pre-division)
  int   counts[4];  // per-class counts
  float invc[4];    // 1/cnt_c (0 if cnt==0)
  float S;          // sum of invc
  int   cstar;      // last class with cnt>0 (defines final same_class_mask)
};

// ---------------------------------------------------------------------------
__global__ void ccel_init(LossParams* p) {
  if (threadIdx.x == 0) {
    p->accum = 0.0f;
    p->counts[0] = 0; p->counts[1] = 0; p->counts[2] = 0; p->counts[3] = 0;
  }
}

// One wave (32 lanes) per row: L2 norm, scale, convert to f16; histogram classes.
__global__ __launch_bounds__(256) void ccel_prep(const float* __restrict__ x,
                                                 const int*   __restrict__ target,
                                                 _Float16*    __restrict__ xn,
                                                 LossParams*  __restrict__ p) {
  const int wave = threadIdx.x >> 5;
  const int lane = threadIdx.x & 31;
  const int row  = blockIdx.x * 8 + wave;

  const float* rp = x + (size_t)row * EDIM + lane * 8;
  const float4 a = ((const float4*)rp)[0];
  const float4 b = ((const float4*)rp)[1];
  float ss = a.x*a.x + a.y*a.y + a.z*a.z + a.w*a.w
           + b.x*b.x + b.y*b.y + b.z*b.z + b.w*b.w;
#pragma unroll
  for (int o = 16; o > 0; o >>= 1) ss += __shfl_xor(ss, o, 32);

  const float scale = 1.0f / fmaxf(sqrtf(ss), NRM_EPS);
  v8h o8;
  o8[0] = (_Float16)(a.x*scale); o8[1] = (_Float16)(a.y*scale);
  o8[2] = (_Float16)(a.z*scale); o8[3] = (_Float16)(a.w*scale);
  o8[4] = (_Float16)(b.x*scale); o8[5] = (_Float16)(b.y*scale);
  o8[6] = (_Float16)(b.z*scale); o8[7] = (_Float16)(b.w*scale);
  *(v8h*)(xn + (size_t)row * EDIM + lane * 8) = o8;

  if (lane == 0) atomicAdd(&p->counts[target[row] & 3], 1);
}

__global__ void ccel_stats(LossParams* p) {
  float S = 0.0f; int cstar = 0;
#pragma unroll
  for (int c = 0; c < 4; ++c) {
    const int cnt = p->counts[c];
    const float inv = (cnt > 0) ? (1.0f / (float)cnt) : 0.0f;
    p->invc[c] = inv;
    S += inv;
    if (cnt > 0) cstar = c;
  }
  p->S = S;
  p->cstar = cstar;
}

// ---------------------------------------------------------------------------
// Fused GEMM + loss. 256 threads = 8 waves (4 wave-rows x 2 wave-cols);
// WG tile 256x128; wave tile 64x64 (4x4 grid of 16x16 WMMA tiles).
// K loop: 8 steps of 32 (f16). 16 WMMA per k-step vs 16 b128 loads.
//
// Fragment addressing (ISA 7.12.2, wave32, 16-bit):
//  A (16x32 MxK): lane L holds row M=L%16; half=L/16 selects K base:
//    v[0..7]  = K in [half*8,  half*8+8)      (contiguous 16B)
//    v[8..15] = K in [16+half*8, 16+half*8+8) (contiguous 16B)
//  B (32x16 KxN): lane L holds col N=L%16; same K split as A:
//    v[0..7]  = K in [half*8,  half*8+8), v[8..15] = K in [16+half*8, 16+half*8+8)
// Since B = Xn^T column-slab, B's column n is just row (colBase+n) of Xn.
// ---------------------------------------------------------------------------
__global__ __launch_bounds__(256, 1) void ccel_gemm(const _Float16* __restrict__ xn,
                                                    const int*      __restrict__ target,
                                                    LossParams*     __restrict__ p) {
  const int bi = blockIdx.y;   // block-row  (256 rows)
  const int bj = blockIdx.x;   // block-col  (128 cols)
  // tile contributes iff its max col index (bj*128+127) exceeds min row (bi*256)
  if (bj < 2 * bi) return;

  const int tid  = threadIdx.x;
  const int wave = tid >> 5;
  const int lane = tid & 31;
  const int half = lane >> 4;
  const int l16  = lane & 15;
  const int waveRow = wave & 3;    // 4 wave-rows of 64
  const int waveCol = wave >> 2;   // 2 wave-cols of 64
  const int rowBase = bi * 256 + waveRow * 64;
  const int colBase = bj * 128 + waveCol * 64;

  v8f acc[4][4] = {};

  const _Float16* aPtr[4];
#pragma unroll
  for (int tm = 0; tm < 4; ++tm)
    aPtr[tm] = xn + (size_t)(rowBase + tm * 16 + l16) * EDIM + half * 8;
  const _Float16* bPtr[4];
#pragma unroll
  for (int tn = 0; tn < 4; ++tn)
    bPtr[tn] = xn + (size_t)(colBase + tn * 16 + l16) * EDIM + half * 8;

  for (int kb = 0; kb < EDIM; kb += KSTEP) {
    v16h A[4], B[4];
#pragma unroll
    for (int tm = 0; tm < 4; ++tm) {
      const v8h lo = *(const v8h*)(aPtr[tm] + kb);
      const v8h hi = *(const v8h*)(aPtr[tm] + kb + 16);
      A[tm] = __builtin_shufflevector(lo, hi, 0,1,2,3,4,5,6,7,8,9,10,11,12,13,14,15);
    }
#pragma unroll
    for (int tn = 0; tn < 4; ++tn) {
      const v8h lo = *(const v8h*)(bPtr[tn] + kb);
      const v8h hi = *(const v8h*)(bPtr[tn] + kb + 16);
      B[tn] = __builtin_shufflevector(lo, hi, 0,1,2,3,4,5,6,7,8,9,10,11,12,13,14,15);
    }
#pragma unroll
    for (int tm = 0; tm < 4; ++tm)
#pragma unroll
      for (int tn = 0; tn < 4; ++tn)
        acc[tm][tn] = __builtin_amdgcn_wmma_f32_16x16x32_f16(
            false, A[tm], false, B[tn], (short)0, acc[tm][tn], false, false);
  }

  // ---- fused loss epilogue -------------------------------------------------
  const float S     = p->S;
  const int   cstar = p->cstar;

  int tj[4]; float invj[4]; bool jstar[4];
#pragma unroll
  for (int tn = 0; tn < 4; ++tn) {
    tj[tn]    = target[colBase + tn * 16 + l16] & 3;
    invj[tn]  = p->invc[tj[tn]];
    jstar[tn] = (tj[tn] == cstar);
  }

  float local = 0.0f;
#pragma unroll
  for (int tm = 0; tm < 4; ++tm) {
#pragma unroll
    for (int r = 0; r < 8; ++r) {
      // C/D layout: element (M = r + 8*half, N = l16) in vgpr r
      const int   i     = rowBase + tm * 16 + half * 8 + r;
      const int   ti    = target[i] & 3;
      const float invi  = p->invc[ti];
      const bool  istar = (ti == cstar);
#pragma unroll
      for (int tn = 0; tn < 4; ++tn) {
        const int j = colBase + tn * 16 + l16;
        if (j > i) {
          const float c = acc[tm][tn][r];
          const float w = (ti == tj[tn]) ? S : (S - invi - invj[tn]);
          float contrib = (1.0f - c) * w;
          if (istar != jstar[tn]) contrib += fmaxf(c - MARGIN, 0.0f);
          local += contrib;
        }
      }
    }
  }

  // wave32 butterfly reduce, then cross-wave via LDS, one atomic per WG
#pragma unroll
  for (int o = 16; o > 0; o >>= 1) local += __shfl_xor(local, o, 32);
  __shared__ float red[8];
  if (lane == 0) red[wave] = local;
  __syncthreads();
  if (tid == 0) {
    float s = 0.0f;
#pragma unroll
    for (int w = 0; w < 8; ++w) s += red[w];
    atomicAdd(&p->accum, s);
  }
}

__global__ void ccel_final(const LossParams* __restrict__ p, float* __restrict__ out) {
  if (threadIdx.x == 0) out[0] = p->accum * (1.0f / (float)N_ROWS);
}

// ---------------------------------------------------------------------------
extern "C" void kernel_launch(void* const* d_in, const int* in_sizes, int n_in,
                              void* d_out, int out_size, void* d_ws, size_t ws_size,
                              hipStream_t stream) {
  const float* x      = (const float*)d_in[0];
  const int*   target = (const int*)d_in[1];

  _Float16*  xn = (_Float16*)d_ws;                                  // 4 MB f16
  LossParams* p = (LossParams*)((char*)d_ws + (size_t)N_ROWS * EDIM * sizeof(_Float16));

  ccel_init <<<dim3(1),          dim3(32),  0, stream>>>(p);
  ccel_prep <<<dim3(N_ROWS / 8), dim3(256), 0, stream>>>(x, target, xn, p);
  ccel_stats<<<dim3(1),          dim3(1),   0, stream>>>(p);
  ccel_gemm <<<dim3(N_ROWS / 128, N_ROWS / 256), dim3(256), 0, stream>>>(xn, target, p);
  ccel_final<<<dim3(1),          dim3(1),   0, stream>>>(p, (float*)d_out);
}
